// ContractiveLoss_90563680404074
// MI455X (gfx1250) — compile-verified, matrix operand order checked by reading, not run on hardware
//
#include <hip/hip_runtime.h>

typedef __attribute__((ext_vector_type(16))) _Float16 v16h;
typedef __attribute__((ext_vector_type(8)))  _Float16 v8h;
typedef __attribute__((ext_vector_type(8)))  float    v8f;
typedef __attribute__((ext_vector_type(4)))  float    v4f;
typedef __attribute__((ext_vector_type(4)))  unsigned int v4u;

static constexpr int   BN   = 8192;   // batch rows (and S is BN x BN)
static constexpr int   DIMK = 256;    // feature dim
static constexpr float MARG = 0.2f;
static constexpr float PDE  = 1e-6f;  // pairwise_distance eps (added to diff)
static constexpr float CEPS = 1e-8f;  // cosine norm clamp

// ---------------------------------------------------------------- accum init
__global__ void init_accum(float* __restrict__ acc) {
    acc[0] = 0.0f;   // triplet sum
    acc[1] = 0.0f;   // sim (sq-diff) sum
}

// ------------------------------------------------ per-row stats + normalize
// One wave per row. Each lane owns 8 contiguous floats (2 x float4 loads).
// Produces: f16 normalized anchor rows (AH), atomic triplet-sum contribution.
__global__ __launch_bounds__(256)
void row_stats(const float* __restrict__ A, const float* __restrict__ P,
               const float* __restrict__ N, _Float16* __restrict__ AH,
               float* __restrict__ accum)
{
    const int lane = threadIdx.x & 31;
    const int row  = (blockIdx.x * blockDim.x + threadIdx.x) >> 5;
    const size_t base = (size_t)row * DIMK + (size_t)lane * 8;

    v4f a0 = *(const v4f*)(A + base);
    v4f a1 = *(const v4f*)(A + base + 4);
    v4f p0 = *(const v4f*)(P + base);
    v4f p1 = *(const v4f*)(P + base + 4);
    v4f n0 = *(const v4f*)(N + base);
    v4f n1 = *(const v4f*)(N + base + 4);

    float sp = 0.f, sn = 0.f, sa = 0.f;
#pragma unroll
    for (int j = 0; j < 4; ++j) {
        float dp0 = a0[j] - p0[j] + PDE;  sp = fmaf(dp0, dp0, sp);
        float dp1 = a1[j] - p1[j] + PDE;  sp = fmaf(dp1, dp1, sp);
        float dn0 = a0[j] - n0[j] + PDE;  sn = fmaf(dn0, dn0, sn);
        float dn1 = a1[j] - n1[j] + PDE;  sn = fmaf(dn1, dn1, sn);
        sa = fmaf(a0[j], a0[j], sa);
        sa = fmaf(a1[j], a1[j], sa);
    }
#pragma unroll
    for (int m = 16; m >= 1; m >>= 1) {
        sp += __shfl_xor(sp, m, 32);
        sn += __shfl_xor(sn, m, 32);
        sa += __shfl_xor(sa, m, 32);
    }

    const float inv = 1.0f / fmaxf(sqrtf(sa), CEPS);
    v8h h;
#pragma unroll
    for (int j = 0; j < 4; ++j) {
        h[j]     = (_Float16)(a0[j] * inv);
        h[4 + j] = (_Float16)(a1[j] * inv);
    }
    *(v8h*)(AH + base) = h;   // 16B store, row-major f16

    if (lane == 0) {
        float tri = fmaxf(sqrtf(sp) - sqrtf(sn) + MARG, 0.0f);
        atomicAdd(accum + 0, tri);
    }
}

// --------------------------------------------------------- fragment loader
// CDNA5 16-bit A fragment (16x32, MxK):
//   lanes 0-15 : M = lane,    K = k0+0..7  and k0+16..23
//   lanes 16-31: M = lane-16, K = k0+8..15 and k0+24..31
// Because we compute Ahat * Ahat^T, the B operand (K x 16 slice of Ahat^T)
// is loaded with the identical pattern using the column-tile's row index.
__device__ inline v16h load_frag(const _Float16* __restrict__ AH,
                                 int row0, int k0, int lane)
{
    const int m  = lane & 15;
    const int ko = (lane >> 4) << 3;        // 0 or 8
    const _Float16* p = AH + (size_t)(row0 + m) * DIMK + (k0 + ko);
    union { v16h h; v4u u[2]; } f;
    f.u[0] = *(const v4u*)(p);              // K = k0+ko   .. +7   (b128)
    f.u[1] = *(const v4u*)(p + 16);         // K = k0+ko+16.. +23  (b128)
    return f.h;
}

// ----------------------------------------------- fused GEMM + MSE epilogue
// One wave computes a 64x64 tile of cos = Ahat*Ahat^T as 4x4 WMMAs,
// then accumulates sum((cos - S)^2) for that tile. Ahat (4MB f16) stays
// L2-resident; S is streamed from HBM exactly once.
__global__ __launch_bounds__(256)
void gemm_mse(const _Float16* __restrict__ AH, const float* __restrict__ S,
              float* __restrict__ accum)
{
    const int lane = threadIdx.x & 31;
    const int t    = blockIdx.x * 8 + (threadIdx.x >> 5);  // wave-tile id
    const int tm   = (t >> 7) << 6;                        // 128 tiles per dim
    const int tn   = (t & 127) << 6;

    v8f acc[4][4];
    const v8f vzero = {};
#pragma unroll
    for (int i = 0; i < 4; ++i)
#pragma unroll
        for (int j = 0; j < 4; ++j)
            acc[i][j] = vzero;

#pragma unroll
    for (int k0 = 0; k0 < DIMK; k0 += 32) {
        v16h af[4];
#pragma unroll
        for (int i = 0; i < 4; ++i)
            af[i] = load_frag(AH, tm + 16 * i, k0, lane);
#pragma unroll
        for (int j = 0; j < 4; ++j) {
            v16h bf = load_frag(AH, tn + 16 * j, k0, lane);
#pragma unroll
            for (int i = 0; i < 4; ++i)
                acc[i][j] = __builtin_amdgcn_wmma_f32_16x16x32_f16(
                    false, af[i], false, bf, (short)0, acc[i][j], false, false);
        }
    }

    // C/D layout: VGPR r -> row = 16*i + 8*(lane>>4) + r ; col = 16*j + (lane&15)
    const int rbase = (lane >> 4) << 3;
    const int col   = lane & 15;
    float local = 0.f;
#pragma unroll
    for (int i = 0; i < 4; ++i) {
#pragma unroll
        for (int j = 0; j < 4; ++j) {
            const size_t s0 = (size_t)(tm + 16 * i + rbase) * BN
                            + (size_t)(tn + 16 * j + col);
#pragma unroll
            for (int r = 0; r < 8; ++r) {
                float d = acc[i][j][r] - S[s0 + (size_t)r * BN];
                local = fmaf(d, d, local);
            }
        }
    }
#pragma unroll
    for (int m = 16; m >= 1; m >>= 1)
        local += __shfl_xor(local, m, 32);
    if (lane == 0)
        atomicAdd(accum + 1, local);
}

// ---------------------------------------------------------------- finalize
__global__ void finalize(const float* __restrict__ accum, float* __restrict__ out)
{
    out[0] = accum[0] * (1.0f / (float)BN)
           + accum[1] * (1.0f / ((float)BN * (float)BN));
}

// ------------------------------------------------------------------ launch
extern "C" void kernel_launch(void* const* d_in, const int* in_sizes, int n_in,
                              void* d_out, int out_size, void* d_ws, size_t ws_size,
                              hipStream_t stream)
{
    (void)in_sizes; (void)n_in; (void)out_size; (void)ws_size;

    const float* A  = (const float*)d_in[0];   // anchor   [8192,256] f32
    const float* P  = (const float*)d_in[1];   // positive [8192,256] f32
    const float* Ng = (const float*)d_in[2];   // negative [8192,256] f32
    const float* S  = (const float*)d_in[3];   // similarity [8192,8192] f32
    float* out = (float*)d_out;

    float*    accum = (float*)d_ws;                        // 2 floats
    _Float16* AH    = (_Float16*)((char*)d_ws + 256);      // 4 MB f16 Ahat

    init_accum<<<1, 1, 0, stream>>>(accum);
    row_stats<<<BN / 8, 256, 0, stream>>>(A, P, Ng, AH, accum);
    gemm_mse<<<(BN / 64) * (BN / 64) / 8, 256, 0, stream>>>(AH, S, accum);
    finalize<<<1, 1, 0, stream>>>(accum, out);
}
